// LlamaAttention_81346680586367
// MI455X (gfx1250) — matrix operand order repaired and checked by reading, not compiled, not run on hardware
//
#include <hip/hip_runtime.h>
#include <math.h>
#include <stdint.h>

typedef __bf16 bf16;
typedef __attribute__((ext_vector_type(16))) __bf16 v16bf;
typedef __attribute__((ext_vector_type(8)))  float  v8f;
typedef __attribute__((ext_vector_type(4)))  unsigned int u32x4;

#define S_LEN 2048
#define HID   2048
#define NQH   32
#define NKVH  8
#define HD    64

union FragBF { v16bf v; u32x4 u[2]; };
union FragF  { v8f   v; float f[8]; };
union Pack8  { u32x4 u; bf16 h[8]; };

// CDNA5 async global->LDS copy (ASYNCcnt-tracked, no VGPR round trip).
// VDST = per-lane LDS byte offset; VADDR = per-lane 64-bit global address.
__device__ __forceinline__ void async_copy_b128(uint32_t lds_off, const void* gsrc) {
    uint64_t ga = (uint64_t)(uintptr_t)gsrc;
    asm volatile("global_load_async_to_lds_b128 %0, %1, off"
                 :: "v"(lds_off), "v"(ga) : "memory");
}
__device__ __forceinline__ void wait_async0() {
    asm volatile("s_wait_asynccnt 0" ::: "memory");
}

// ---------------------------------------------------------------------------
// fp32 -> bf16 elementwise convert
// ---------------------------------------------------------------------------
__global__ void f32_to_bf16(const float* __restrict__ in, bf16* __restrict__ out, int n) {
    int i = blockIdx.x * blockDim.x + threadIdx.x;
    if (i < n) out[i] = (bf16)in[i];
}

// ---------------------------------------------------------------------------
// Tiled WMMA GEMM: C[M,N] = A[M,K] @ B[K,N], bf16 in, fp32 accumulate.
// Block tile 256x64, K-step 32. 8 waves; each wave: 32 rows x 64 cols
// (2 A-frags reuse 4 B-frags -> 8 WMMAs per K-step per wave).
// MODE 0: bf16 out, head layout [n0/64][S][64]           (V projection)
// MODE 1: bf16 out, head layout, fused RoPE              (Q/K projections)
// MODE 2: fp32 out, row-major [M][N]                     (output projection)
// ---------------------------------------------------------------------------
template <int MODE>
__global__ __launch_bounds__(256) void gemm_wmma(
    const bf16* __restrict__ A, const bf16* __restrict__ B, void* __restrict__ C,
    const int* __restrict__ pos_ids, int M, int N, int K)
{
    const int tid  = threadIdx.x;
    const int lane = tid & 31, wave = tid >> 5;
    const int half = lane >> 4, l16 = lane & 15;
    const int m0 = blockIdx.x * 256;
    const int n0 = blockIdx.y * 64;

    __shared__ __align__(16) bf16 As[256][40];   // [m][k], padded rows (80B)
    __shared__ __align__(16) bf16 Bst[64][40];   // [n][k] transposed, padded

    v8f acc[2][4];
    for (int c = 0; c < 2; ++c)
        for (int g = 0; g < 4; ++g)
            for (int j = 0; j < 8; ++j) acc[c][g][j] = 0.0f;

    for (int k0 = 0; k0 < K; k0 += 32) {
        // --- stage A tile: 256x32 bf16 via async copy (one 64B row per thread)
        async_copy_b128((uint32_t)(uintptr_t)&As[tid][0],
                        A + (size_t)(m0 + tid) * K + k0);
        async_copy_b128((uint32_t)(uintptr_t)&As[tid][8],
                        A + (size_t)(m0 + tid) * K + k0 + 8);
        async_copy_b128((uint32_t)(uintptr_t)&As[tid][16],
                        A + (size_t)(m0 + tid) * K + k0 + 16);
        async_copy_b128((uint32_t)(uintptr_t)&As[tid][24],
                        A + (size_t)(m0 + tid) * K + k0 + 24);
        // --- stage B tile transposed: 32x64 bf16, 8 threads/row, b128 loads
        {
            int r = tid >> 3, cs = (tid & 7) * 8;
            Pack8 p; p.u = *(const u32x4*)(B + (size_t)(k0 + r) * N + n0 + cs);
            for (int j = 0; j < 8; ++j) Bst[cs + j][r] = p.h[j];
        }
        wait_async0();
        __syncthreads();

        // --- 4 B fragments (32x16 each): elems e -> k=half*16+e, n=g*16+l16
        FragBF bfr[4];
        for (int g = 0; g < 4; ++g) {
            const int bn = g * 16 + l16;
            bfr[g].u[0] = *(const u32x4*)&Bst[bn][half * 16];
            bfr[g].u[1] = *(const u32x4*)&Bst[bn][half * 16 + 8];
        }
        // --- 2 A fragments, 8 WMMAs
        for (int c = 0; c < 2; ++c) {
            FragBF af;
            const int ar = wave * 32 + c * 16 + l16;
            af.u[0] = *(const u32x4*)&As[ar][half * 16];
            af.u[1] = *(const u32x4*)&As[ar][half * 16 + 8];
            for (int g = 0; g < 4; ++g)
                acc[c][g] = __builtin_amdgcn_wmma_f32_16x16x32_bf16(
                    false, af.v, false, bfr[g].v, (short)0, acc[c][g], false, false);
        }
        __syncthreads();
    }

    // --- epilogue: elem e -> row m0+wave*32+c*16+half*8+e, col n0+g*16+l16
    if (MODE == 2) {
        float* out = (float*)C;
        for (int c = 0; c < 2; ++c)
            for (int g = 0; g < 4; ++g) {
                FragF f; f.v = acc[c][g];
                int n = n0 + g * 16 + l16;
                for (int e = 0; e < 8; ++e) {
                    int m = m0 + wave * 32 + c * 16 + half * 8 + e;
                    out[(size_t)m * N + n] = f.f[e];
                }
            }
    } else {
        bf16* out = (bf16*)C;
        const int head = n0 >> 6;  // BN == head_dim == 64
        for (int c = 0; c < 2; ++c) {
            FragF f[4];
            for (int g = 0; g < 4; ++g) f[g].v = acc[c][g];
            for (int g = 0; g < 4; ++g) {
                int d = g * 16 + l16;
                float invf = 0.0f;
                if (MODE == 1)  // inv_freq = theta^(-(d%32)/32), ln(5e5)=13.122363
                    invf = __expf(-(float)(d & 31) * (1.0f / 32.0f) * 13.122363378f);
                for (int e = 0; e < 8; ++e) {
                    int m = m0 + wave * 32 + c * 16 + half * 8 + e;
                    float v = f[g].f[e];
                    if (MODE == 1) {
                        float ang = (float)pos_ids[m] * invf;
                        float cc, ss;
                        __sincosf(ang, &ss, &cc);
                        float vr = (d < 32) ? -f[g ^ 2].f[e] : f[g ^ 2].f[e];
                        v = v * cc + vr * ss;
                    }
                    out[((size_t)head * S_LEN + m) * HD + d] = (bf16)v;
                }
            }
        }
    }
}

// ---------------------------------------------------------------------------
// Flash attention, causal, GQA (4 Q heads per KV head).
// Grid: (32 heads, 16 q-tiles of 128). 8 waves/block, 16 q rows per wave.
// KV tile 64: 8 score WMMAs + 8 PV WMMAs per softmax round per wave.
// K tile staged once per WG in LDS via CDNA5 async copies (ASYNCcnt); score
// fragments then come from low-latency ds_load_b128 instead of per-wave
// redundant VMEM. V transposed through LDS; P reshaped C->A per-wave.
// ---------------------------------------------------------------------------
__global__ __launch_bounds__(256) void attn_wmma(
    const bf16* __restrict__ Q,   // [32][S][64]
    const bf16* __restrict__ Km,  // [8][S][64]
    const bf16* __restrict__ V,   // [8][S][64]
    bf16* __restrict__ O)         // [S][2048] head-major rows
{
    const int tid  = threadIdx.x;
    const int lane = tid & 31, wave = tid >> 5;
    const int half = lane >> 4, l16 = lane & 15;
    const int h  = blockIdx.x;
    const int kh = h >> 2;
    const int qt = blockIdx.y;
    const int qbase = qt * 128 + wave * 16;

    __shared__ __align__(16) bf16 Ks[64][72];      // [r][d] K tile, row-major
    __shared__ __align__(16) bf16 Vt[64][72];      // [d][r] transposed V tile
    __shared__ __align__(16) bf16 Ps[8][16][72];   // per-wave P tile 16x64

    // Q fragments: 2 chunks along D (k = c*32 + half*16 + e)
    FragBF qf[2];
    {
        const bf16* qp = Q + ((size_t)h * S_LEN + qbase + l16) * HD;
        for (int c = 0; c < 2; ++c) {
            qf[c].u[0] = *(const u32x4*)(qp + c * 32 + half * 16);
            qf[c].u[1] = *(const u32x4*)(qp + c * 32 + half * 16 + 8);
        }
    }

    v8f o[4];
    for (int g = 0; g < 4; ++g)
        for (int j = 0; j < 8; ++j) o[g][j] = 0.0f;
    float rmax[8], rsum[8];
    for (int e = 0; e < 8; ++e) { rmax[e] = -1e30f; rsum[e] = 0.0f; }

    const float scale = 0.125f;            // 1/sqrt(64)
    const int kv_end = qt * 128 + 128;     // WG-uniform causal bound

    for (int kv0 = 0; kv0 < kv_end; kv0 += 64) {
        const int r = tid >> 2, cs = (tid & 3) * 16;
        // K tile (64x64) -> LDS via async copies, overlapped with V staging
        {
            const bf16* kpg = Km + ((size_t)kh * S_LEN + kv0 + r) * HD + cs;
            async_copy_b128((uint32_t)(uintptr_t)&Ks[r][cs],     kpg);
            async_copy_b128((uint32_t)(uintptr_t)&Ks[r][cs + 8], kpg + 8);
        }
        // V tile (64x64) transposed into LDS through VGPRs
        {
            const bf16* vp = V + ((size_t)kh * S_LEN + kv0 + r) * HD + cs;
            Pack8 p0, p1;
            p0.u = *(const u32x4*)(vp);
            p1.u = *(const u32x4*)(vp + 8);
            for (int j = 0; j < 8; ++j) {
                Vt[cs + j][r]     = p0.h[j];
                Vt[cs + 8 + j][r] = p1.h[j];
            }
        }
        wait_async0();
        __syncthreads();

        // scores S = Q @ K^T : four 16x16 kv tiles, accumulated over 2 D-chunks
        v8f sacc[4];
        for (int g = 0; g < 4; ++g)
            for (int j = 0; j < 8; ++j) sacc[g][j] = 0.0f;
        for (int c = 0; c < 2; ++c) {
            for (int g = 0; g < 4; ++g) {
                FragBF kf;   // B frag: n = kv col, elem e -> d = c*32+half*16+e
                kf.u[0] = *(const u32x4*)&Ks[g * 16 + l16][c * 32 + half * 16];
                kf.u[1] = *(const u32x4*)&Ks[g * 16 + l16][c * 32 + half * 16 + 8];
                sacc[g] = __builtin_amdgcn_wmma_f32_16x16x32_bf16(
                    false, qf[c].v, false, kf.v, (short)0, sacc[g], false, false);
            }
        }

        // online softmax in C-fragment layout
        FragF s[4];
        for (int g = 0; g < 4; ++g) s[g].v = sacc[g];
        float tmax[8];
        for (int e = 0; e < 8; ++e) {
            int row = qbase + half * 8 + e;
            float m01, m23;
            {
                float a = s[0].f[e] * scale;
                float b = s[1].f[e] * scale;
                if (kv0 + l16 > row)      a = -1e30f;
                if (kv0 + 16 + l16 > row) b = -1e30f;
                s[0].f[e] = a; s[1].f[e] = b;
                m01 = fmaxf(a, b);
            }
            {
                float a = s[2].f[e] * scale;
                float b = s[3].f[e] * scale;
                if (kv0 + 32 + l16 > row) a = -1e30f;
                if (kv0 + 48 + l16 > row) b = -1e30f;
                s[2].f[e] = a; s[3].f[e] = b;
                m23 = fmaxf(a, b);
            }
            tmax[e] = fmaxf(m01, m23);
        }
        for (int m = 1; m < 16; m <<= 1)
            for (int e = 0; e < 8; ++e)
                tmax[e] = fmaxf(tmax[e], __shfl_xor(tmax[e], m, 32));
        float alpha[8];
        for (int e = 0; e < 8; ++e) {
            float nm = fmaxf(rmax[e], tmax[e]);
            alpha[e] = __expf(rmax[e] - nm);
            rmax[e] = nm;
            for (int g = 0; g < 4; ++g)
                s[g].f[e] = __expf(s[g].f[e] - nm);
        }
        float tsum[8];
        for (int e = 0; e < 8; ++e)
            tsum[e] = (s[0].f[e] + s[1].f[e]) + (s[2].f[e] + s[3].f[e]);
        for (int m = 1; m < 16; m <<= 1)
            for (int e = 0; e < 8; ++e)
                tsum[e] += __shfl_xor(tsum[e], m, 32);
        for (int e = 0; e < 8; ++e) rsum[e] = rsum[e] * alpha[e] + tsum[e];
        for (int g = 0; g < 4; ++g) {
            FragF of; of.v = o[g];
            for (int e = 0; e < 8; ++e) of.f[e] *= alpha[e];
            o[g] = of.v;
        }

        // P: C layout -> A layout via per-wave LDS slice (in-order DS per wave)
        for (int g = 0; g < 4; ++g)
            for (int e = 0; e < 8; ++e)
                Ps[wave][half * 8 + e][g * 16 + l16] = (bf16)s[g].f[e];
        FragBF pf[2];
        for (int c = 0; c < 2; ++c) {
            pf[c].u[0] = *(const u32x4*)&Ps[wave][l16][c * 32 + half * 16];
            pf[c].u[1] = *(const u32x4*)&Ps[wave][l16][c * 32 + half * 16 + 8];
        }

        // O += P @ V : 2 kv-chunks x 4 d-groups = 8 WMMAs
        for (int c = 0; c < 2; ++c) {
            for (int g = 0; g < 4; ++g) {
                FragBF vf;   // B frag: n = d, elem e -> kv = c*32+half*16+e
                vf.u[0] = *(const u32x4*)&Vt[g * 16 + l16][c * 32 + half * 16];
                vf.u[1] = *(const u32x4*)&Vt[g * 16 + l16][c * 32 + half * 16 + 8];
                o[g] = __builtin_amdgcn_wmma_f32_16x16x32_bf16(
                    false, pf[c].v, false, vf.v, (short)0, o[g], false, false);
            }
        }
        __syncthreads();
    }

    // epilogue: divide by row sum, write bf16 head-major rows
    for (int g = 0; g < 4; ++g) {
        FragF of; of.v = o[g];
        int d = g * 16 + l16;
        for (int e = 0; e < 8; ++e) {
            int row = qbase + half * 8 + e;
            O[(size_t)row * HID + h * HD + d] = (bf16)(of.f[e] / rsum[e]);
        }
    }
}

// ---------------------------------------------------------------------------
extern "C" void kernel_launch(void* const* d_in, const int* in_sizes, int n_in,
                              void* d_out, int out_size, void* d_ws, size_t ws_size,
                              hipStream_t stream) {
    const float* hs  = (const float*)d_in[0];
    const int*   pos = (const int*)d_in[1];
    const float* Wq  = (const float*)d_in[2];
    const float* Wk  = (const float*)d_in[3];
    const float* Wv  = (const float*)d_in[4];
    const float* Wo  = (const float*)d_in[5];

    char* ws = (char*)d_ws;
    size_t off = 0;
    auto alloc = [&](size_t bytes) {
        char* p = ws + off;
        off += (bytes + 255) & ~(size_t)255;
        return p;
    };
    const int KVD = NKVH * HD;  // 512
    bf16* Xb  = (bf16*)alloc((size_t)S_LEN * HID * 2);
    bf16* Wqb = (bf16*)alloc((size_t)HID * HID * 2);
    bf16* Wkb = (bf16*)alloc((size_t)HID * KVD * 2);
    bf16* Wvb = (bf16*)alloc((size_t)HID * KVD * 2);
    bf16* Wob = (bf16*)alloc((size_t)HID * HID * 2);
    bf16* Qb  = (bf16*)alloc((size_t)NQH * S_LEN * HD * 2);
    bf16* Kb  = (bf16*)alloc((size_t)NKVH * S_LEN * HD * 2);
    bf16* Vb  = (bf16*)alloc((size_t)NKVH * S_LEN * HD * 2);
    bf16* Ab  = (bf16*)alloc((size_t)S_LEN * HID * 2);

    auto cvt = [&](const float* src, bf16* dst, int n) {
        f32_to_bf16<<<(n + 255) / 256, 256, 0, stream>>>(src, dst, n);
    };
    cvt(hs, Xb, S_LEN * HID);
    cvt(Wq, Wqb, HID * HID);
    cvt(Wk, Wkb, HID * KVD);
    cvt(Wv, Wvb, HID * KVD);
    cvt(Wo, Wob, HID * HID);

    dim3 blk(256);
    // Q/K projections with fused RoPE, head-layout bf16 output
    gemm_wmma<1><<<dim3(S_LEN / 256, HID / 64), blk, 0, stream>>>(
        Xb, Wqb, Qb, pos, S_LEN, HID, HID);
    gemm_wmma<1><<<dim3(S_LEN / 256, KVD / 64), blk, 0, stream>>>(
        Xb, Wkb, Kb, pos, S_LEN, KVD, HID);
    // V projection, head layout
    gemm_wmma<0><<<dim3(S_LEN / 256, KVD / 64), blk, 0, stream>>>(
        Xb, Wvb, Vb, pos, S_LEN, KVD, HID);
    // flash attention
    attn_wmma<<<dim3(NQH, S_LEN / 128), blk, 0, stream>>>(Qb, Kb, Vb, Ab);
    // output projection -> fp32 d_out
    gemm_wmma<2><<<dim3(S_LEN / 256, HID / 64), blk, 0, stream>>>(
        Ab, Wob, (float*)d_out, pos, S_LEN, HID, HID);
}
